// ShiftedPearsonLoss_1614907703983
// MI455X (gfx1250) — compile-verified, matrix operand order checked by reading, not run on hardware
//
#include <hip/hip_runtime.h>
#include <math.h>

typedef __attribute__((ext_vector_type(16))) _Float16 v16h;
typedef __attribute__((ext_vector_type(8)))  float    v8f;

#define D_LEN   4096
#define S_SH    64
#define W_SH    129          // 2*S+1 shifts
#define NJ      136          // K-steps padded to 17 per wave (132 real + 4 zero)
#define NSTEPS  17           // NJ / NWAVES, uniform per wave
#define XPAD_LEN 4368        // max A index = 32*135+31+15 = 4366
#define YOFF    240          // front pad so B index >= 0 for n<=15
#define YPAD_LEN 4592        // max B index = 32*135+31+240 = 4591
#define NWAVES  8
#define BLOCK   256

__device__ __forceinline__ float block_reduce_sum(float v, float* red) {
  const int tid = threadIdx.x;
  red[tid] = v; __syncthreads();
  for (int s = BLOCK / 2; s > 0; s >>= 1) {
    if (tid < s) red[tid] += red[tid + s];
    __syncthreads();
  }
  float r = red[0]; __syncthreads();
  return r;
}

__device__ __forceinline__ float block_reduce_min(float v, float* red) {
  const int tid = threadIdx.x;
  red[tid] = v; __syncthreads();
  for (int s = BLOCK / 2; s > 0; s >>= 1) {
    if (tid < s) red[tid] = fminf(red[tid], red[tid + s]);
    __syncthreads();
  }
  float r = red[0]; __syncthreads();
  return r;
}

__device__ __forceinline__ float block_reduce_max(float v, float* red) {
  const int tid = threadIdx.x;
  red[tid] = v; __syncthreads();
  for (int s = BLOCK / 2; s > 0; s >>= 1) {
    if (tid < s) red[tid] = fmaxf(red[tid], red[tid + s]);
    __syncthreads();
  }
  float r = red[0]; __syncthreads();
  return r;
}

// Gather one 16x32 f16 A fragment (shifted-x Toeplitz rows) and one 32x16
// f16 B fragment (lag-shifted centered-y columns) for K-step j.
// Per-lane layout follows ISA 7.12.2 (16-bit A 16x32): element e <-> K index
// k = 8h + (e&7) + ((e&8)<<1), h = lane>>4. The two 8-element runs are
// contiguous in LDS, so these lower to 2x ds_load_b128 per fragment; with j
// a compile-time-constant step from a uniform base, the step contribution
// becomes an immediate ds_load offset (no per-step address VALU).
__device__ __forceinline__ void load_frags(const _Float16* __restrict__ xph,
                                           const _Float16* __restrict__ yph,
                                           int j, int ln, int h,
                                           v16h& av, v16h& bv) {
  const int base = 32 * j + h * 8;
#pragma unroll
  for (int e = 0; e < 16; ++e) {
    const int k = ((e & 8) << 1) | (e & 7);
    av[e] = xph[base + k + ln];                // A[m][k] = xpad[32j+k+m]
    bv[e] = yph[base + k - 16 * ln + YOFF];    // B[k][n] = ypad[32j+k-16n]
  }
}

__device__ __forceinline__ v8f wmma_step(const v16h& a, const v16h& b, v8f c) {
  return __builtin_amdgcn_wmma_f32_16x16x32_f16(
      /*neg_a=*/false, a, /*neg_b=*/false, b,
      /*c_mod=*/(short)0, c, /*reuse_a=*/false, /*reuse_b=*/false);
}

// One workgroup per (b, c) pair. Computes max over 129 shifts of the
// Pearson correlation between the Hann-windowed normalized (flipped-channel)
// x row and the y row. Numerator Sxy for all shifts is computed with
// v_wmma_f32_16x16x32_f16 via a Toeplitz tiling:
//   C[m][n] = sum_j sum_k xpad[32j+k+m] * ypad[32j+k-16n+YOFF] = Sxy[16n+m]
__global__ __launch_bounds__(BLOCK) void pair_corr_kernel(
    const float* __restrict__ x, const float* __restrict__ y,
    float* __restrict__ pairvals) {
  __shared__ _Float16 xph[XPAD_LEN];         // padded windowed x (f16)
  __shared__ _Float16 yph[YPAD_LEN];         // padded centered y (f16)
  __shared__ float    xw32[D_LEN];           // windowed x (f32) for edge sums
  __shared__ float    red[BLOCK];            // reduction scratch
  __shared__ float    cbuf[NWAVES][BLOCK];   // per-wave partial C tiles
  __shared__ float    corrArr[W_SH];

  const int tid = threadIdx.x;
  const int p = blockIdx.x;
  const int b = p >> 1, c = p & 1;
  const float* __restrict__ xr = x + ((size_t)b * 2 + (size_t)(1 - c)) * D_LEN; // flip axis 1
  const float* __restrict__ yr = y + ((size_t)b * 2 + (size_t)c) * D_LEN;

  // ---- row min / max of x ----
  float vmin = INFINITY, vmax = -INFINITY;
  for (int d = tid; d < D_LEN; d += BLOCK) {
    float v = xr[d];
    vmin = fminf(vmin, v);
    vmax = fmaxf(vmax, v);
  }
  const float xmin = block_reduce_min(vmin, red);
  const float xmax = block_reduce_max(vmax, red);
  const float scale = 2.0f / (xmax - xmin);

  // ---- windowed normalized x; totals Sx_tot, Sxx_tot ----
  const float w0 = 6.28318530717958647692f / (float)D_LEN;
  float ls = 0.0f, ls2 = 0.0f;
  for (int d = tid; d < D_LEN; d += BLOCK) {
    float v = (xr[d] - xmin) * scale - 1.0f;
    float hann = 0.5f * (1.0f - __cosf(w0 * (float)d));
    v *= hann;
    xw32[d] = v;
    xph[S_SH + d] = (_Float16)v;
    ls += v;
    ls2 += v * v;
  }
  for (int i = tid; i < S_SH; i += BLOCK) xph[i] = (_Float16)0.0f;
  for (int i = S_SH + D_LEN + tid; i < XPAD_LEN; i += BLOCK) xph[i] = (_Float16)0.0f;
  const float sx_tot  = block_reduce_sum(ls, red);
  const float sxx_tot = block_reduce_sum(ls2, red);

  // ---- centered y; Syy ----
  float ys = 0.0f;
  for (int d = tid; d < D_LEN; d += BLOCK) ys += yr[d];
  const float ybar = block_reduce_sum(ys, red) * (1.0f / (float)D_LEN);
  float ss = 0.0f;
  for (int d = tid; d < D_LEN; d += BLOCK) {
    float v = yr[d] - ybar;
    yph[YOFF + d] = (_Float16)v;
    ss += v * v;
  }
  for (int i = tid; i < YOFF; i += BLOCK) yph[i] = (_Float16)0.0f;
  for (int i = YOFF + D_LEN + tid; i < YPAD_LEN; i += BLOCK) yph[i] = (_Float16)0.0f;
  const float syy = block_reduce_sum(ss, red);
  __syncthreads();

  // ---- WMMA K-loop: 136 steps of 16x16x32 f16, exactly 17 per wave.
  // Fully unrolled (constant trip count), double-buffered fragments, and
  // TWO accumulators so consecutive WMMAs have no accumulator RAW hazard
  // (kills the mandated v_nop spacing and lets the matrix pipe stream).
  const int lane = tid & 31;
  const int wave = tid >> 5;
  const int ln = lane & 15;   // A: row m   B: col n
  const int h  = lane >> 4;   // lane half selects K sub-block
  v8f acc0 = {}, acc1 = {};
  {
    v16h a0, b0, a1, b1;
    load_frags(xph, yph, wave, ln, h, a0, b0);
#pragma unroll
    for (int t = 0; t < NSTEPS - 1; t += 2) {
      load_frags(xph, yph, wave + (t + 1) * NWAVES, ln, h, a1, b1);
      acc0 = wmma_step(a0, b0, acc0);
      load_frags(xph, yph, wave + (t + 2) * NWAVES, ln, h, a0, b0);
      acc1 = wmma_step(a1, b1, acc1);
    }
    acc0 = wmma_step(a0, b0, acc0);   // step 16
  }
  v8f acc = acc0 + acc1;
#pragma unroll
  for (int i = 0; i < 8; ++i) cbuf[wave][lane * 8 + i] = acc[i];
  __syncthreads();

  // ---- reduce partial tiles, per-shift stats, correlation ----
  {
    const int l = tid >> 3, i = tid & 7;   // slot (lane l, vgpr i) of C tile
    float sxy = 0.0f;
#pragma unroll
    for (int wv = 0; wv < NWAVES; ++wv) sxy += cbuf[wv][l * 8 + i];
    const int n = l & 15;
    const int m = i + 8 * (l >> 4);        // 32-bit C/D 16x16 VGPR layout
    const int wsh = 16 * n + m;
    if (wsh < W_SH) {
      // window [wsh, wsh+D) of xpad differs from full support [S, S+D) only
      // at <=64-element edges: subtract the excluded pieces from the totals.
      float sx = sx_tot, sxx = sxx_tot;
      for (int d = 0; d < wsh - S_SH; ++d) {             // left exclusion (wsh > S)
        float v = xw32[d]; sx -= v; sxx -= v * v;
      }
      for (int d = wsh + D_LEN - S_SH; d < D_LEN; ++d) { // right exclusion (wsh < S)
        float v = xw32[d]; sx -= v; sxx -= v * v;
      }
      const float varx = sxx - sx * sx * (1.0f / (float)D_LEN);
      corrArr[wsh] = sxy * rsqrtf(varx * syy);
    }
  }
  __syncthreads();

  float cm = (tid < W_SH) ? corrArr[tid] : -INFINITY;
  const float best = block_reduce_max(cm, red);
  if (tid == 0) pairvals[p] = best;
}

// scores[b] = sqrt(v[b,0]^2 + v[b,1]^2); out = -mean(scores)
__global__ __launch_bounds__(64) void finalize_kernel(
    const float* __restrict__ pairvals, float* __restrict__ out) {
  __shared__ float red[64];
  const int t = threadIdx.x;
  float v0 = pairvals[2 * t + 0];
  float v1 = pairvals[2 * t + 1];
  red[t] = sqrtf(v0 * v0 + v1 * v1);
  __syncthreads();
  for (int s = 32; s > 0; s >>= 1) {
    if (t < s) red[t] += red[t + s];
    __syncthreads();
  }
  if (t == 0) out[0] = -red[0] * (1.0f / 64.0f);
}

extern "C" void kernel_launch(void* const* d_in, const int* in_sizes, int n_in,
                              void* d_out, int out_size, void* d_ws, size_t ws_size,
                              hipStream_t stream) {
  const float* x = (const float*)d_in[0];   // (64, 2, 4096) f32
  const float* y = (const float*)d_in[1];   // (64, 2, 4096) f32
  float* ws = (float*)d_ws;                 // 128 floats: per-(b,c) max corr
  pair_corr_kernel<<<128, BLOCK, 0, stream>>>(x, y, ws);
  finalize_kernel<<<1, 64, 0, stream>>>(ws, (float*)d_out);
}
